// PolicyHead_20332375179986
// MI455X (gfx1250) — matrix operand order repaired and checked by reading, not compiled, not run on hardware
//
#include <hip/hip_runtime.h>
#include <math.h>

typedef __attribute__((ext_vector_type(8)))  float  v8f;
typedef __bf16 bf16;
typedef __attribute__((ext_vector_type(16))) __bf16 v16bf;
typedef __attribute__((ext_vector_type(8)))  __bf16 v8bf;

#define HW   361
#define CIN  512
#define CP1  256
#define CG1  256
#define NOUT 6
#define NB   256
#define LDSK 40   // padded row stride (bf16 elems): 80B, 16B-aligned, bank-conflict-free

__device__ __forceinline__ float mishf(float x) {
  float sp = (x > 20.f) ? x : log1pf(__expf(x));
  return x * tanhf(sp);
}

__device__ __forceinline__ void atomicMaxF(float* addr, float val) {
  int* ia = (int*)addr;
  int old = __float_as_int(*addr);
  while (__int_as_float(old) < val) {
    int assumed = old;
    old = atomicCAS(ia, assumed, __float_as_int(val));
    if (old == assumed) break;
  }
}

// ---------------------------------------------------------------------------
// Prep: pack [conv1p_w; conv1g_w] (512 x 512 fp32) into bf16 A-fragments laid
// out per lane for V_WMMA_F32_16X16X32_BF16 in contiguous k order (as B), and init the
// cross-workgroup sum/max reduction buffers.
// A 16-bit 16x32 layout: lane = m%16 + 16*(khalf); VGPR v holds K pair:
//   k = v*2 + (lane>=16 ? 16 : 0)  (contiguous: k = 16*khalf + s)
// Fragment storage: Apack[((mtile*16 + ktile)*32 + lane)*8 + v] (dword pairs)
// ---------------------------------------------------------------------------
__global__ __launch_bounds__(256) void k_prep(const float* __restrict__ w1p,
                                              const float* __restrict__ w1g,
                                              bf16* __restrict__ Apack,
                                              float* __restrict__ ws_sum,
                                              float* __restrict__ ws_max) {
  const int idx = blockIdx.x * 256 + threadIdx.x;
  if (idx < 131072) {
    const int v     = idx & 7;
    const int lane  = (idx >> 3) & 31;
    const int ktile = (idx >> 8) & 15;
    const int mtile = idx >> 12;
    const int k = ktile * 32 + ((lane & 16) ? 16 : 0) + v * 2;
    const int m = mtile * 16 + (lane & 15);
    const float* wr = (m < CP1) ? (w1p + (size_t)m * CIN)
                                : (w1g + (size_t)(m - CP1) * CIN);
    Apack[idx * 2]     = (bf16)wr[k];
    Apack[idx * 2 + 1] = (bf16)wr[k + 1];
  } else if (idx < 131072 + 65536) {
    ws_sum[idx - 131072] = 0.f;
  } else if (idx < 131072 + 131072) {
    ws_max[idx - 196608] = -1e30f;
  }
}

// mask_sum per batch: one wave32 per batch, shuffle reduce
__global__ void k_masksum(const float* __restrict__ mask, float* __restrict__ ws_ms) {
  const int b = blockIdx.x, lane = threadIdx.x;
  float s = 0.f;
  for (int i = lane; i < HW; i += 32) s += mask[b * HW + i];
  #pragma unroll
  for (int o = 16; o > 0; o >>= 1) s += __shfl_xor(s, o);
  if (lane == 0) ws_ms[b] = s;
}

// ---------------------------------------------------------------------------
// Fused GEMM: Y[512 x 361] = Wcat[512 x 512] @ x[b][512 x 361]  (bf16 WMMA)
// Block tile: 256(M) x 64(N); 8 waves as 4(M) x 2(N); wave = 64M x 32N
// m-chunk 0 -> conv1p pre-activation (store fp32); m-chunk 1 -> conv1g path
// with fused mish + masked mean/max pooling (shuffle-reduced, then atomics).
// ---------------------------------------------------------------------------
__global__ __launch_bounds__(256) void k_gemm1(
    const float* __restrict__ x, const float* __restrict__ mask,
    const float* __restrict__ beta_g, const bf16* __restrict__ Apack,
    float* __restrict__ ws_outp, float* __restrict__ ws_sum,
    float* __restrict__ ws_max)
{
  __shared__ bf16 ldsb[64 * LDSK];         // x tile, n-major: [n][k], 5 KB
  const int nc = blockIdx.x;               // 0..5  (n chunk of 64)
  const int mchunk = blockIdx.y;           // 0..1  (0 = p path, 1 = g path)
  const int b = blockIdx.z;
  const int tid = threadIdx.x;
  const int lane = tid & 31, wave = tid >> 5;
  const int wave_m = wave & 3, wave_n = wave >> 2;
  const int n0 = nc * 64;

  const float* xb = x + (size_t)b * CIN * HW;

  v8f acc[4][2];
  #pragma unroll
  for (int i = 0; i < 4; ++i)
    #pragma unroll
    for (int j = 0; j < 2; ++j) acc[i][j] = (v8f)(0.f);

  const int n_local = tid & 63;
  const int kbase   = (tid >> 6) << 3;     // 0,8,16,24
  const int hw_ld   = n0 + n_local;
  // Clamp the address so staging loads are UNPREDICATED (no exec-mask
  // branch + serialized waits in the hot loop); zero OOB lanes with a
  // cheap v_cndmask after the load. Clamped element is real, finite data.
  const bool  inb_ld  = hw_ld < HW;
  const int   hw_safe = inb_ld ? hw_ld : (HW - 1);

  for (int ks = 0; ks < 16; ++ks) {
    const int c0 = ks * 32;
    // --- stage x tile: coalesced fp32 loads (one clause), convert, b128 store
    float xr[8];
    #pragma unroll
    for (int j = 0; j < 8; ++j) {
      float v = xb[(size_t)(c0 + kbase + j) * HW + hw_safe];
      xr[j] = inb_ld ? v : 0.f;
    }
    __syncthreads();
    v8bf pk;
    #pragma unroll
    for (int j = 0; j < 8; ++j) pk[j] = (bf16)xr[j];
    *(v8bf*)(&ldsb[n_local * LDSK + kbase]) = pk;
    __syncthreads();

    // --- A fragments: pre-swizzled, 2x global_load_b128 per fragment ---
    v16bf afr[4];
    #pragma unroll
    for (int i = 0; i < 4; ++i) {
      const int mtile = mchunk * 16 + wave_m * 4 + i;
      afr[i] = *(const v16bf*)(Apack + ((((mtile * 16 + ks) * 32) + lane) << 4));
    }
    if (ks + 1 < 16) {
      const int mtile = mchunk * 16 + wave_m * 4;
      __builtin_prefetch(Apack + ((((mtile * 16 + ks + 1) * 32) + lane) << 4), 0, 1);
    }

    // --- B fragments from LDS (contiguous 32B per lane) + 8 WMMAs ---
    #pragma unroll
    for (int j = 0; j < 2; ++j) {
      const int n  = wave_n * 32 + j * 16 + (lane & 15);
      const int k0 = (lane & 16) ? 16 : 0;
      v16bf bfr = *(const v16bf*)(&ldsb[n * LDSK + k0]);
      #pragma unroll
      for (int i = 0; i < 4; ++i)
        acc[i][j] = __builtin_amdgcn_wmma_f32_16x16x32_bf16(
            false, afr[i], false, bfr, (short)0, acc[i][j], false, false);
    }
  }

  // --- epilogue: C/D layout lane = ncol (+16 for m+8 rows), VGPR v = m row ---
  const int ncol  = lane & 15;
  const int mhalf = (lane & 16) ? 8 : 0;
  #pragma unroll
  for (int j = 0; j < 2; ++j) {
    const int hw = n0 + wave_n * 32 + j * 16 + ncol;
    const bool inb = hw < HW;
    const int  hws = inb ? hw : (HW - 1);
    const float mraw = mask[b * HW + hws];        // unpredicated load
    const float m = inb ? mraw : 0.f;
    #pragma unroll
    for (int i = 0; i < 4; ++i) {
      #pragma unroll
      for (int v = 0; v < 8; ++v) {
        const int ch = wave_m * 64 + i * 16 + mhalf + v;   // 0..255 in chunk
        if (mchunk == 0) {
          if (inb) ws_outp[((size_t)b * CP1 + ch) * HW + hw] = acc[i][j][v];
        } else {
          float val = inb ? mishf((acc[i][j][v] + beta_g[ch]) * m) : 0.f;
          float s   = inb ? val * m : 0.f;
          float mx  = inb ? (val + m - 1.f) : -1e30f;
          #pragma unroll
          for (int o = 8; o > 0; o >>= 1) {       // reduce the 16 hw lanes
            s += __shfl_xor(s, o);
            mx = fmaxf(mx, __shfl_xor(mx, o));
          }
          if (ncol == 0) {
            atomicAdd(&ws_sum[b * CG1 + ch], s);
            atomicMaxF(&ws_max[b * CG1 + ch], mx);
          }
        }
      }
    }
  }
}

// ---------------------------------------------------------------------------
// Gate: pooled(768) -> g(256) per batch. pooled built in LDS from sum/max.
// ---------------------------------------------------------------------------
__global__ __launch_bounds__(256) void k_gate(
    const float* __restrict__ Lw, const float* __restrict__ ws_sum,
    const float* __restrict__ ws_max, const float* __restrict__ ws_ms,
    float* __restrict__ ws_g)
{
  __shared__ float pooled[768];
  const int b = blockIdx.x, t = threadIdx.x;
  const float ms   = ws_ms[b];
  const float off  = (sqrtf(ms) - 14.f) * 0.1f;
  const float mean = ws_sum[b * CG1 + t] / ms;
  pooled[t]       = mean;
  pooled[256 + t] = mean * off;
  pooled[512 + t] = ws_max[b * CG1 + t];
  __syncthreads();
  float a = 0.f;
  const float* lw = Lw + (size_t)t * 768;
  #pragma unroll 8
  for (int j = 0; j < 768; ++j) a = fmaf(pooled[j], lw[j], a);
  ws_g[b * CP1 + t] = a;
}

// ---------------------------------------------------------------------------
// Final: out[b][o][hw] = sum_c w2[o][c]*mish((outp+g+b2)*mask) - (1-mask)*5000
// ---------------------------------------------------------------------------
__global__ __launch_bounds__(192) void k_final(
    const float* __restrict__ ws_outp, const float* __restrict__ ws_g,
    const float* __restrict__ beta2, const float* __restrict__ w2,
    const float* __restrict__ mask, float* __restrict__ out)
{
  const int b  = blockIdx.y;
  const int hw = blockIdx.x * 192 + threadIdx.x;
  if (hw >= HW) return;
  const float m = mask[b * HW + hw];
  float acc[NOUT];
  #pragma unroll
  for (int o = 0; o < NOUT; ++o) acc[o] = 0.f;
  const float* op = ws_outp + (size_t)b * CP1 * HW + hw;
  const float* gb = ws_g + (size_t)b * CP1;
  for (int c = 0; c < CP1; ++c) {
    const float val = mishf((op[(size_t)c * HW] + gb[c] + beta2[c]) * m);
    #pragma unroll
    for (int o = 0; o < NOUT; ++o) acc[o] = fmaf(w2[o * CP1 + c], val, acc[o]);
  }
  #pragma unroll
  for (int o = 0; o < NOUT; ++o)
    out[((size_t)b * NOUT + o) * HW + hw] = acc[o] - (1.f - m) * 5000.f;
}

// ---------------------------------------------------------------------------
extern "C" void kernel_launch(void* const* d_in, const int* in_sizes, int n_in,
                              void* d_out, int out_size, void* d_ws, size_t ws_size,
                              hipStream_t stream) {
  (void)in_sizes; (void)n_in; (void)out_size; (void)ws_size;
  const float* x    = (const float*)d_in[0];
  const float* mask = (const float*)d_in[1];
  const float* w1p  = (const float*)d_in[2];
  const float* w1g  = (const float*)d_in[3];
  const float* bg   = (const float*)d_in[4];
  const float* Lw   = (const float*)d_in[5];
  const float* b2   = (const float*)d_in[6];
  const float* w2   = (const float*)d_in[7];
  float* out = (float*)d_out;

  // workspace layout
  float* ws_outp = (float*)d_ws;                      // 256*256*361 fp32
  bf16*  Apack   = (bf16*)(ws_outp + 23658496);       // 262144 bf16 (512 KB)
  float* ws_sum  = (float*)(Apack + 262144);          // 256*256
  float* ws_max  = ws_sum + 65536;                    // 256*256
  float* ws_ms   = ws_max + 65536;                    // 256
  float* ws_g    = ws_ms + 256;                       // 256*256

  k_prep<<<1024, 256, 0, stream>>>(w1p, w1g, Apack, ws_sum, ws_max);
  k_masksum<<<NB, 32, 0, stream>>>(mask, ws_ms);
  k_gemm1<<<dim3(6, 2, NB), 256, 0, stream>>>(x, mask, bg, Apack, ws_outp,
                                              ws_sum, ws_max);
  k_gate<<<NB, 256, 0, stream>>>(Lw, ws_sum, ws_max, ws_ms, ws_g);
  k_final<<<dim3(2, NB), 192, 0, stream>>>(ws_outp, ws_g, b2, w2, mask, out);
}
